// WindowAttention_26792005992990
// MI455X (gfx1250) — compile-verified, matrix operand order checked by reading, not run on hardware
//
#include <hip/hip_runtime.h>

typedef __attribute__((ext_vector_type(8)))  _Float16 v8h;
typedef __attribute__((ext_vector_type(16))) _Float16 v16h;
typedef __attribute__((ext_vector_type(8)))  float    v8f;

// ---------------------------------------------------------------------------
// Fragment load from a row-major f16 LDS tile (stride in elements).
// A/B fragment layout (16x32 f16): lane&15 = row, lane&16 ? K=8..15,24..31 : K=0..7,16..23
// ---------------------------------------------------------------------------
__device__ __forceinline__ v16h ld_frag(const _Float16* base, int stride, int lane) {
  const _Float16* p = base + (lane & 15) * stride + ((lane & 16) ? 8 : 0);
  v8h lo = *(const v8h*)(p);
  v8h hi = *(const v8h*)(p + 16);
  v16h f;
#pragma unroll
  for (int i = 0; i < 8; ++i) { f[i] = lo[i]; f[i + 8] = hi[i]; }
  return f;
}

__device__ __forceinline__ v8f wmma32(v16h a, v16h b, v8f c) {
  return __builtin_amdgcn_wmma_f32_16x16x32_f16(false, a, false, b, (short)0, c, false, false);
}

// ---- register staging for the double-buffered GEMM pipeline ---------------
struct RegsF { float4 v[4]; };
struct RegsH { v8h v[2]; };

__device__ __forceinline__ RegsF greg(const float* __restrict__ s) {
  RegsF r; const float4* s4 = (const float4*)s;
#pragma unroll
  for (int i = 0; i < 4; ++i) r.v[i] = s4[i];
  return r;
}
__device__ __forceinline__ RegsH greg(const _Float16* __restrict__ s) {
  RegsH r; r.v[0] = *(const v8h*)s; r.v[1] = *(const v8h*)(s + 8); return r;
}
__device__ __forceinline__ void s2l(const RegsF& r, _Float16* d) {
#pragma unroll
  for (int i = 0; i < 4; ++i) {
    d[4*i+0] = (_Float16)r.v[i].x; d[4*i+1] = (_Float16)r.v[i].y;
    d[4*i+2] = (_Float16)r.v[i].z; d[4*i+3] = (_Float16)r.v[i].w;
  }
}
__device__ __forceinline__ void s2l(const RegsH& r, _Float16* d) {
  *(v8h*)d = r.v[0]; *(v8h*)(d + 8) = r.v[1];
}

// ---------------------------------------------------------------------------
// Prep: W (K x N, f32 row-major) -> Wt (N x K, f16 row-major)
// ---------------------------------------------------------------------------
__global__ void transpose_f16_kernel(const float* __restrict__ W, _Float16* __restrict__ Wt,
                                     int K, int N) {
  int idx = blockIdx.x * 256 + threadIdx.x;
  if (idx < K * N) {
    int n = idx / K, k = idx % K;
    Wt[idx] = (_Float16)W[(size_t)k * N + n];
  }
}

// Prep: biasm[h][q][k] = bias_table[rel_index[q*49+k]][h]    (6 x 49 x 49 f32)
__global__ void biasmat_kernel(const float* __restrict__ bt, const int* __restrict__ rel,
                               float* __restrict__ out) {
  int idx = blockIdx.x * 256 + threadIdx.x;
  if (idx < 6 * 2401) {
    int h = idx / 2401, qk = idx % 2401;
    out[idx] = bt[rel[qk] * 6 + h];
  }
}

// ---------------------------------------------------------------------------
// Tiled WMMA GEMM: C[M,N] = A[M,K] @ Bt[N,K]^T + bias[N]
// 256 threads = 8 waves; block tile 128x128; each wave owns 32(M) x 64(N).
// Double-buffered LDS K-loop (32 per step). Grid: x = N blocks (fast, A-tile
// L2 reuse), y = M blocks.  M%128==0, N%128==0, K%32==0 (holds here).
// SCALE: multiply columns n < scale_cols by `scale` (folds q-scaling into QKV).
// ---------------------------------------------------------------------------
template <typename AT, bool OUT_F16, bool SCALE>
__global__ __launch_bounds__(256) void gemm_wmma_kernel(
    const AT* __restrict__ A, const _Float16* __restrict__ Bt,
    const float* __restrict__ bias, void* __restrict__ Cout,
    int M, int N, int K, float scale, int scale_cols) {
  __shared__ __attribute__((aligned(16))) _Float16 lA[2][128 * 32];
  __shared__ __attribute__((aligned(16))) _Float16 lB[2][128 * 32];

  const int tid  = threadIdx.x;
  const int lane = tid & 31;
  const int wave = tid >> 5;
  const int wr   = wave >> 1;   // 0..3 : 32-row strip
  const int wc   = wave & 1;    // 0..1 : 64-col strip
  const long n0  = (long)blockIdx.x * 128;   // N fastest -> A-tile reused in L2
  const long m0  = (long)blockIdx.y * 128;

  v8f acc[2][4];
#pragma unroll
  for (int t = 0; t < 2; ++t)
#pragma unroll
    for (int u = 0; u < 4; ++u)
#pragma unroll
      for (int j = 0; j < 8; ++j) acc[t][u][j] = 0.f;

  const int row  = tid >> 1;          // 0..127
  const int colb = (tid & 1) * 16;    // 0 or 16
  const AT*       aptr = A  + (m0 + row) * (long)K + colb;
  const _Float16* bptr = Bt + (n0 + row) * (long)K + colb;
  const int ldd = row * 32 + colb;

  // stage 0
  { auto ra = greg(aptr); auto rb = greg(bptr);
    s2l(ra, &lA[0][ldd]); s2l(rb, &lB[0][ldd]); }
  __syncthreads();

  const int NK = K >> 5;
  for (int it = 0; it < NK; ++it) {
    const int cur = it & 1, nxt = cur ^ 1;
    decltype(greg(aptr)) ra;
    RegsH rb;
    const bool has = (it + 1) < NK;
    if (has) { ra = greg(aptr + (it + 1) * 32); rb = greg(bptr + (it + 1) * 32); }

    v16h af[2], bf[4];
#pragma unroll
    for (int t = 0; t < 2; ++t) af[t] = ld_frag(&lA[cur][(wr * 32 + t * 16) * 32], 32, lane);
#pragma unroll
    for (int u = 0; u < 4; ++u) bf[u] = ld_frag(&lB[cur][(wc * 64 + u * 16) * 32], 32, lane);
#pragma unroll
    for (int t = 0; t < 2; ++t)
#pragma unroll
      for (int u = 0; u < 4; ++u) acc[t][u] = wmma32(af[t], bf[u], acc[t][u]);

    if (has) { s2l(ra, &lA[nxt][ldd]); s2l(rb, &lB[nxt][ldd]); }
    __syncthreads();
  }

  // Epilogue: C/D layout -> vgpr j = m%8, lane = n + 16*(m/8)
#pragma unroll
  for (int t = 0; t < 2; ++t)
#pragma unroll
    for (int u = 0; u < 4; ++u) {
      int  mloc = wr * 32 + t * 16 + ((lane >> 4) << 3);
      long n_g  = n0 + wc * 64 + u * 16 + (lane & 15);
      float bv  = bias[n_g];
      float sc  = (SCALE && n_g < scale_cols) ? scale : 1.f;
#pragma unroll
      for (int j = 0; j < 8; ++j) {
        long m_g = m0 + mloc + j;
        float v  = (acc[t][u][j] + bv) * sc;
        if (OUT_F16) ((_Float16*)Cout)[m_g * N + n_g] = (_Float16)v;
        else         ((float*)Cout)[m_g * N + n_g]    = v;
      }
    }
}

// ---------------------------------------------------------------------------
// Fused window attention: one workgroup (128 thr = 4 waves) per (window b, head h).
// S=49 padded to 64. qkv: f16 [B_*49, 1152] with Q pre-scaled. Output f16 [B_*49, 384].
// ---------------------------------------------------------------------------
__global__ __launch_bounds__(128) void attn_kernel(
    const _Float16* __restrict__ qkv, const float* __restrict__ mask,
    const float* __restrict__ biasm, _Float16* __restrict__ attout) {
  __shared__ __attribute__((aligned(16))) _Float16 qt[64 * 64];  // Q, later out staging
  __shared__ __attribute__((aligned(16))) _Float16 kt[64 * 64];
  __shared__ __attribute__((aligned(16))) _Float16 vt[64 * 64];  // V^T : [e][k]
  __shared__ __attribute__((aligned(16))) _Float16 pt[64 * 64];  // softmax probs
  __shared__ float sc[64 * 64];                                  // bias+mask, then scores

  const int tid  = threadIdx.x;
  const int lane = tid & 31;
  const int wave = tid >> 5;           // 0..3 -> 16-row strip of queries
  const int bh   = blockIdx.x;
  const int b    = bh / 6;
  const int h    = bh % 6;
  const int wm   = b & 63;             // mask window index

  // zero pad regions (rows/cols 49..63 must be 0 in q/k/v tiles)
  v8h z;
#pragma unroll
  for (int j = 0; j < 8; ++j) z[j] = (_Float16)0.f;
  for (int i = tid; i < 512; i += 128) {
    ((v8h*)qt)[i] = z; ((v8h*)kt)[i] = z; ((v8h*)vt)[i] = z;
  }
  __syncthreads();

  // Load Q,K rows, V transposed, and pre-stage (bias + mask) into sc.
  // Thread t<98: row s = t>>1, half = (t&1)*32.  t>=98: fill sc pad rows.
  if (tid < 98) {
    const int s = tid >> 1, half = (tid & 1) * 32;
    const _Float16* base = qkv + ((size_t)(b * 49 + s)) * 1152 + h * 64 + half;
#pragma unroll
    for (int i = 0; i < 4; ++i) {
      *(v8h*)(qt + s * 64 + half + 8 * i) = *(const v8h*)(base + 8 * i);
      *(v8h*)(kt + s * 64 + half + 8 * i) = *(const v8h*)(base + 384 + 8 * i);
    }
    const _Float16* vsrc = base + 768;
#pragma unroll
    for (int e = 0; e < 32; ++e) vt[(half + e) * 64 + s] = vsrc[e];

    // bias+mask for row s, cols half..half+31 (contiguous global reads)
    const float* bmr = biasm + h * 2401 + s * 49;
    const float* mkr = mask  + wm * 2401 + s * 49;
    float* dst = sc + s * 64 + half;
#pragma unroll
    for (int c = 0; c < 32; ++c) {
      int col = half + c;
      dst[c] = (col < 49) ? (bmr[col] + mkr[col]) : -1e30f;
    }
  } else {
    const int s2 = tid - 98;                       // 0..29 -> pad rows 49..63
    const int q = 49 + (s2 >> 1), half = (s2 & 1) * 32;
    float* dst = sc + q * 64 + half;
#pragma unroll
    for (int c = 0; c < 32; ++c) dst[c] = -1e30f;
  }
  __syncthreads();

  // attn = Q @ K^T  (64x64, K-dim 64 in 2 steps)
  v8f a4[4];
#pragma unroll
  for (int u = 0; u < 4; ++u)
#pragma unroll
    for (int j = 0; j < 8; ++j) a4[u][j] = 0.f;
#pragma unroll
  for (int k0 = 0; k0 < 64; k0 += 32) {
    v16h a = ld_frag(qt + (wave * 16) * 64 + k0, 64, lane);
#pragma unroll
    for (int u = 0; u < 4; ++u) {
      v16h bb = ld_frag(kt + (u * 16) * 64 + k0, 64, lane);
      a4[u] = wmma32(a, bb, a4[u]);
    }
  }
  // scores += pre-staged bias+mask (branch-free LDS read-modify-write;
  // pad cells hold -1e30 and pad accumulators are 0)
#pragma unroll
  for (int u = 0; u < 4; ++u) {
    int kk = u * 16 + (lane & 15);
    int qb = wave * 16 + ((lane >> 4) << 3);
#pragma unroll
    for (int j = 0; j < 8; ++j) {
      float* cell = &sc[(qb + j) * 64 + kk];
      *cell = a4[u][j] + *cell;
    }
  }
  __syncthreads();

  // row softmax over 49 valid keys; write f16 probs (pad cols = 0)
  if (tid < 64) {
    float* r = sc + tid * 64;
    float mx = -1e30f;
    for (int c = 0; c < 49; ++c) mx = fmaxf(mx, r[c]);
    float sum = 0.f;
    for (int c = 0; c < 49; ++c) { float e = __expf(r[c] - mx); r[c] = e; sum += e; }
    float inv = 1.f / sum;
    _Float16* p = pt + tid * 64;
    for (int c = 0; c < 49; ++c) p[c] = (_Float16)(r[c] * inv);
    for (int c = 49; c < 64; ++c) p[c] = (_Float16)0.f;
  }
  __syncthreads();

  // out = P @ V   (B fragment = V^T rows, n = e)
  v8f o4[4];
#pragma unroll
  for (int u = 0; u < 4; ++u)
#pragma unroll
    for (int j = 0; j < 8; ++j) o4[u][j] = 0.f;
#pragma unroll
  for (int k0 = 0; k0 < 64; k0 += 32) {
    v16h a = ld_frag(pt + (wave * 16) * 64 + k0, 64, lane);
#pragma unroll
    for (int u = 0; u < 4; ++u) {
      v16h bb = ld_frag(vt + (u * 16) * 64 + k0, 64, lane);
      o4[u] = wmma32(a, bb, o4[u]);
    }
  }

  // repack output through LDS (qt is free) -> 64B vector global stores
  {
    int qb = wave * 16 + ((lane >> 4) << 3);
#pragma unroll
    for (int u = 0; u < 4; ++u) {
      int e = u * 16 + (lane & 15);
#pragma unroll
      for (int j = 0; j < 8; ++j) qt[(qb + j) * 64 + e] = (_Float16)o4[u][j];
    }
  }
  __syncthreads();
  if (tid < 98) {
    const int s = tid >> 1, half = (tid & 1) * 32;
    _Float16* dst = attout + ((size_t)(b * 49 + s)) * 384 + h * 64 + half;
    const _Float16* src = qt + s * 64 + half;
#pragma unroll
    for (int i = 0; i < 4; ++i) *(v8h*)(dst + 8 * i) = *(const v8h*)(src + 8 * i);
  }
}

// ---------------------------------------------------------------------------
extern "C" void kernel_launch(void* const* d_in, const int* in_sizes, int n_in,
                              void* d_out, int out_size, void* d_ws, size_t ws_size,
                              hipStream_t stream) {
  (void)in_sizes; (void)n_in; (void)out_size; (void)ws_size;
  const float* x       = (const float*)d_in[0];
  const float* mask    = (const float*)d_in[1];
  const float* qkv_w   = (const float*)d_in[2];
  const float* qkv_b   = (const float*)d_in[3];
  const float* proj_w  = (const float*)d_in[4];
  const float* proj_b  = (const float*)d_in[5];
  const float* bias_tb = (const float*)d_in[6];
  const int*   rel_idx = (const int*)d_in[7];
  float*       out     = (float*)d_out;

  const int B_ = 4096, S = 49, D = 384, H = 6;
  const int M  = B_ * S;                 // 200704 = 1568 * 128 exactly
  const float scale = 0.125f;            // hd=64 -> 64^-0.5

  // workspace carve-up (256B aligned)
  char*  ws  = (char*)d_ws;
  size_t off = 0;
  auto carve = [&](size_t bytes) {
    void* p = ws + off;
    off = (off + bytes + 255) & ~(size_t)255;
    return p;
  };
  _Float16* qkvWt  = (_Float16*)carve((size_t)3 * D * D * sizeof(_Float16)); // 1152x384
  _Float16* projWt = (_Float16*)carve((size_t)D * D * sizeof(_Float16));     // 384x384
  float*    biasm  = (float*)carve((size_t)H * S * S * sizeof(float));
  _Float16* qkvF   = (_Float16*)carve((size_t)M * 3 * D * sizeof(_Float16)); // ~462 MB
  _Float16* attF   = (_Float16*)carve((size_t)M * D * sizeof(_Float16));     // ~154 MB

  // prep
  transpose_f16_kernel<<<(D * 3 * D + 255) / 256, 256, 0, stream>>>(qkv_w, qkvWt, D, 3 * D);
  transpose_f16_kernel<<<(D * D + 255) / 256, 256, 0, stream>>>(proj_w, projWt, D, D);
  biasmat_kernel<<<(H * S * S + 255) / 256, 256, 0, stream>>>(bias_tb, rel_idx, biasm);

  // QKV GEMM: f32 x @ Wt -> f16 qkv, q columns (n<384) pre-scaled
  gemm_wmma_kernel<float, true, true><<<dim3((3 * D) / 128, M / 128), 256, 0, stream>>>(
      x, qkvWt, qkv_b, qkvF, M, 3 * D, D, scale, D);

  // fused window attention per (b, h)
  attn_kernel<<<B_ * H, 128, 0, stream>>>(qkvF, mask, biasm, attF);

  // projection GEMM: f16 attF @ projWt -> f32 out
  gemm_wmma_kernel<_Float16, false, false><<<dim3(D / 128, M / 128), 256, 0, stream>>>(
      attF, projWt, proj_b, out, M, D, D, 1.f, 0);
}
